// RustGraph_7327214207536
// MI455X (gfx1250) — compile-verified, matrix operand order checked by reading, not run on hardware
//
#include <hip/hip_runtime.h>
#include <math.h>

typedef _Float16 f16;
typedef __attribute__((ext_vector_type(4)))  _Float16 v4h;
typedef __attribute__((ext_vector_type(16))) _Float16 v16h;
typedef __attribute__((ext_vector_type(8)))  float    v8f;

#define ACT_NONE     0
#define ACT_RELU     1
#define ACT_SIGMOID  2
#define ACT_TANH     3
#define ACT_SOFTPLUS 4

template<int ACT>
__device__ __forceinline__ float act_t(float v) {
    if constexpr (ACT == ACT_RELU)     return fmaxf(v, 0.f);
    else if constexpr (ACT == ACT_SIGMOID)  return 1.f / (1.f + expf(-v));
    else if constexpr (ACT == ACT_TANH)     return tanhf(v);
    else if constexpr (ACT == ACT_SOFTPLUS) return v > 20.f ? v : log1pf(expf(v));
    else return v;
}

// ---------------------------------------------------------------------------
// WMMA GEMM:  out[:, oOff:oOff+64] = ACT( A[N,KT] @ W[Kw,64] + bias )
//   A contiguous, row stride KT (zero-padded beyond real K); KT/ACT/HAS2 are
//   compile-time -> fully unrolled K-loop, branch-free epilogue.
// One wave -> 16x64 tile, 4 accumulators, v_wmma_f32_16x16x32_f16.
// afrag is wave-private: same-wave DS ops complete in order, so only a
// zero-cost wave_barrier (compiler fence) is needed per K-step — waves run
// decoupled. Staging: 4 outstanding global_load_b128, then cvt_pk + b64 store.
// ---------------------------------------------------------------------------
template<int KT, int ACT, bool HAS2>
__global__ __launch_bounds__(256)
void k_gemmT(const float* __restrict__ A, int Kw,
             const float* __restrict__ W,
             const float* __restrict__ bias,
             float* __restrict__ out,  int oStr,  int oOff,
             float* __restrict__ out2, int o2Str, int o2Off,
             int N)
{
    constexpr int NK = KT / 32;
    __shared__ __attribute__((aligned(32))) f16 wfrag[NK * 2048]; // [kstep][g][lane*16+j]
    __shared__ __attribute__((aligned(32))) f16 afrag[8 * 512];   // [wave][lane*16+j]

    const int tid     = threadIdx.x;
    const int wave    = tid >> 5;
    const int lane    = tid & 31;
    const int rowBase = blockIdx.x * 128 + wave * 16;

    // Stage weights (f32 -> f16) into fragment order, zero rows Kw..KT.
    for (int idx = tid; idx < KT * 64; idx += 256) {
        int k = idx >> 6, n = idx & 63;
        float w = (k < Kw) ? W[k * 64 + n] : 0.f;
        int kl = k & 31;
        int j  = (kl & 7) | ((kl >> 4) << 3);
        int l  = (n & 15) | (((kl >> 3) & 1) << 4);
        wfrag[(k >> 5) * 2048 + (n >> 4) * 512 + l * 16 + j] = (f16)w;
    }
    __syncthreads();   // wfrag is read-only from here; afrag is wave-private

    v8f cacc[4] = {};
    f16* aw = &afrag[wave * 512];

#pragma unroll
    for (int ks = 0; ks < NK; ++ks) {
        const int kk = ks << 5;

        // Phase 1: issue all 4 tile loads (distinct regs -> 4 in flight).
        float4 vv[4];
#pragma unroll
        for (int i = 0; i < 4; ++i) {
            int t   = lane + i * 32;
            int m   = t >> 3;
            int kq  = (t & 7) << 2;               // multiple of 4
            int row = rowBase + m; row = (row < N) ? row : (N - 1);
            vv[i] = *(const float4*)&A[(long)row * KT + kk + kq];
        }
        // Phase 2: convert + fragment-order LDS stores.
#pragma unroll
        for (int i = 0; i < 4; ++i) {
            int t  = lane + i * 32;
            int m  = t >> 3;
            int kq = (t & 7) << 2;
            int j  = (kq & 7) | ((kq >> 4) << 3);
            int l  = m | (((kq >> 3) & 1) << 4);
            v4h pk;
            pk.x = (f16)vv[i].x; pk.y = (f16)vv[i].y;
            pk.z = (f16)vv[i].z; pk.w = (f16)vv[i].w;
            *(v4h*)&aw[l * 16 + j] = pk;           // 8B-aligned ds_store_b64
        }
        __builtin_amdgcn_wave_barrier();           // order stores before loads

        v16h a = *(const v16h*)&aw[lane * 16];
        const f16* wb = &wfrag[ks * 2048];
#pragma unroll
        for (int g = 0; g < 4; ++g) {
            v16h b = *(const v16h*)&wb[g * 512 + lane * 16];
            cacc[g] = __builtin_amdgcn_wmma_f32_16x16x32_f16(
                          false, a, false, b, (short)0, cacc[g], false, false);
        }
        __builtin_amdgcn_wave_barrier();           // keep next stage behind reads
    }

    // Epilogue: C/D layout — col n = lane&15 (+16g), row m = r + 8*(lane>>4).
    const int n0 = lane & 15;
    const int mh = (lane >> 4) * 8;
    float bcol[4];
#pragma unroll
    for (int g = 0; g < 4; ++g) bcol[g] = bias[g * 16 + n0];

#pragma unroll
    for (int r = 0; r < 8; ++r) {
        int row = rowBase + mh + r;
        if (row < N) {
#pragma unroll
            for (int g = 0; g < 4; ++g) {
                float v = act_t<ACT>(cacc[g][r] + bcol[g]);
                out[(long)row * oStr + oOff + g * 16 + n0] = v;
                if constexpr (HAS2)
                    out2[(long)row * o2Str + o2Off + g * 16 + n0] = v;
            }
        }
    }
}

// ---------------------------------------------------------------------------
// Graph kernels
// ---------------------------------------------------------------------------
__global__ void k_deg(const int* __restrict__ dst, float* __restrict__ deg, int E) {
    int e = blockIdx.x * blockDim.x + threadIdx.x;
    if (e < E) atomicAdd(&deg[dst[e]], 1.0f);
}

__global__ void k_node_norm(const float* __restrict__ deg,
                            float* __restrict__ dinv,
                            float* __restrict__ norms, int N) {
    int i = blockIdx.x * blockDim.x + threadIdx.x;
    if (i < N) {
        float d = deg[i] + 1.f;
        dinv[i]  = rsqrtf(d);
        norms[i] = 1.f / d;
    }
}

__global__ void k_edge_norm(const int* __restrict__ src, const int* __restrict__ dst,
                            const float* __restrict__ dinv,
                            float* __restrict__ norme, int E) {
    int e = blockIdx.x * blockDim.x + threadIdx.x;
    if (e < E) norme[e] = dinv[src[e]] * dinv[dst[e]];
}

// 16 lanes per edge, float4 per lane: gather hW[src] row (L2-resident),
// scatter-add to agg[dst] with global_atomic_add_f32.
__global__ void k_agg(const float* __restrict__ hW,
                      const int* __restrict__ src, const int* __restrict__ dst,
                      const float* __restrict__ norme,
                      float* __restrict__ agg, int E) {
    long t = (long)blockIdx.x * blockDim.x + threadIdx.x;
    int e = (int)(t >> 4), lane = (int)(t & 15);
    if (e >= E) return;
    int s = src[e], d = dst[e];
    float ne = norme[e];
    const float4 v = *(const float4*)(hW + (long)s * 64 + lane * 4);
    float* o = agg + (long)d * 64 + lane * 4;
    atomicAdd(o + 0, v.x * ne);
    atomicAdd(o + 1, v.y * ne);
    atomicAdd(o + 2, v.z * ne);
    atomicAdd(o + 3, v.w * ne);
}

// out = ACT( agg + hW * norm_s + bias [+ accum] )   (float4 over N*16)
template<int ACT>
__global__ void k_finalize(float* __restrict__ out,
                           const float* __restrict__ agg,
                           const float* __restrict__ hW,
                           const float* __restrict__ norms,
                           const float* __restrict__ bias,
                           const float* __restrict__ accum,
                           int N) {
    long idx = (long)blockIdx.x * blockDim.x + threadIdx.x;
    if (idx >= (long)N * 16) return;
    int node = (int)(idx >> 4), q = (int)(idx & 15);
    float  ns = norms[node];
    float4 av = ((const float4*)agg)[idx];
    float4 hv = ((const float4*)hW)[idx];
    float4 bv = ((const float4*)bias)[q];
    float4 r;
    r.x = av.x + hv.x * ns + bv.x;
    r.y = av.y + hv.y * ns + bv.y;
    r.z = av.z + hv.z * ns + bv.z;
    r.w = av.w + hv.w * ns + bv.w;
    if (accum) {
        float4 cv = ((const float4*)accum)[idx];
        r.x += cv.x; r.y += cv.y; r.z += cv.z; r.w += cv.w;
    }
    r.x = act_t<ACT>(r.x); r.y = act_t<ACT>(r.y);
    r.z = act_t<ACT>(r.z); r.w = act_t<ACT>(r.w);
    ((float4*)out)[idx] = r;
}

// ---------------------------------------------------------------------------
// Packing + elementwise (float4)
// ---------------------------------------------------------------------------
__global__ void k_pack64(const float* __restrict__ src, int ss, int soff,
                         float* __restrict__ dst, int ds, int doff, int N) {
    long idx = (long)blockIdx.x * blockDim.x + threadIdx.x;
    if (idx >= (long)N * 16) return;
    int node = (int)(idx >> 4), q = (int)(idx & 15);
    float4 v = *(const float4*)(src + (long)node * ss + soff + q * 4);
    *(float4*)(dst + (long)node * ds + doff + q * 4) = v;
}

// catP[N,96] = [ h1 | diff | zeros ]
__global__ void k_packP(const float* __restrict__ h1, const float* __restrict__ diff,
                        float* __restrict__ catP, int N) {
    long idx = (long)blockIdx.x * blockDim.x + threadIdx.x;
    if (idx >= (long)N * 24) return;
    int node = (int)(idx / 24), q = (int)(idx % 24);
    float4 r = make_float4(0.f, 0.f, 0.f, 0.f);
    if (q < 16)       r = *(const float4*)(h1 + (long)node * 64 + q * 4);
    else if (q == 16) r.x = diff[node];
    *(float4*)(catP + (long)node * 96 + q * 4) = r;
}

__global__ void k_zcomb(const float* __restrict__ eps, const float* __restrict__ es,
                        const float* __restrict__ em, float* __restrict__ z, long n4) {
    long i = (long)blockIdx.x * blockDim.x + threadIdx.x;
    if (i >= n4) return;
    float4 a = ((const float4*)eps)[i], s = ((const float4*)es)[i], m = ((const float4*)em)[i];
    float4 r; r.x = a.x*s.x + m.x; r.y = a.y*s.y + m.y; r.z = a.z*s.z + m.z; r.w = a.w*s.w + m.w;
    ((float4*)z)[i] = r;
}
__global__ void k_mul(const float* __restrict__ a, const float* __restrict__ b,
                      float* __restrict__ o, long n4) {
    long i = (long)blockIdx.x * blockDim.x + threadIdx.x;
    if (i >= n4) return;
    float4 x = ((const float4*)a)[i], y = ((const float4*)b)[i];
    float4 r; r.x = x.x*y.x; r.y = x.y*y.y; r.z = x.z*y.z; r.w = x.w*y.w;
    ((float4*)o)[i] = r;
}
__global__ void k_gru(const float* __restrict__ zg, const float* __restrict__ hi,
                      const float* __restrict__ ht, float* __restrict__ o, long n4) {
    long i = (long)blockIdx.x * blockDim.x + threadIdx.x;
    if (i >= n4) return;
    float4 z = ((const float4*)zg)[i], h = ((const float4*)hi)[i], t = ((const float4*)ht)[i];
    float4 r;
    r.x = z.x*h.x + (1.f-z.x)*t.x; r.y = z.y*h.y + (1.f-z.y)*t.y;
    r.z = z.z*h.z + (1.f-z.z)*t.z; r.w = z.w*h.w + (1.f-z.w)*t.w;
    ((float4*)o)[i] = r;
}

// ---------------------------------------------------------------------------
// Host orchestration
// ---------------------------------------------------------------------------
extern "C" void kernel_launch(void* const* d_in, const int* in_sizes, int n_in,
                              void* d_out, int out_size, void* d_ws, size_t ws_size,
                              hipStream_t stream) {
    const float* x    = (const float*)d_in[0];     // [N,128]
    const float* hbuf = (const float*)d_in[1];     // [2,N,64]
    const float* diff = (const float*)d_in[2];     // [N,1]
    const int*   eidx = (const int*)d_in[3];       // [2,E]
    const float* eps  = (const float*)d_in[4];     // [N,64]

    const int N = in_sizes[0] / 128;
    const int E = in_sizes[3] / 2;
    const long N64 = (long)N * 64;
    const long N16 = (long)N * 16;

    int p = 5;
    auto nxt = [&]() { return (const float*)d_in[p++]; };
    const float *Wpx = nxt(), *bpx = nxt();
    const float *Wpz = nxt(), *bpz = nxt();
    const float *Wen = nxt(), *ben = nxt();
    const float *Wem = nxt(), *bem = nxt();
    const float *Wes = nxt(), *bes = nxt();
    const float *Wpr = nxt(), *bpr = nxt();
    const float *Wpm = nxt(), *bpm = nxt();
    const float *Wps = nxt(), *bps = nxt();
    const float *Wg[2][6]; const float *bg[2][6];  // xz,hz,xr,hr,xh,hh
    for (int l = 0; l < 2; ++l)
        for (int g = 0; g < 6; ++g) { Wg[l][g] = nxt(); bg[l][g] = nxt(); }

    const int* src = eidx;
    const int* dst = eidx + E;
    const float* h0 = hbuf;
    const float* h1 = hbuf + N64;

    // Workspace carve-up (all slices 16B aligned)
    float* ws     = (float*)d_ws;
    float* deg    = ws;            ws += N;
    float* dinv   = ws;            ws += N;
    float* norms  = ws;            ws += N;
    float* zero64 = ws;            ws += 64;             // zero bias slab
    float* norme  = ws;            ws += E;
    float* catXH  = ws;            ws += (long)N * 128;  // [phiX | h1]
    float* catXZ  = ws;            ws += (long)N * 128;  // [phiX | phiZ]
    float* catP   = ws;            ws += (long)N * 96;   // [h1 | diff | 0]
    float* tmpHW  = ws;            ws += N64;
    float* agg    = ws;            ws += N64;
    float* encx   = ws;            ws += N64;
    float* priorx = ws;            ws += N64;
    float* tmp1   = ws;            ws += N64;
    float* zgb    = ws;            ws += N64;
    float* rgb    = ws;            ws += N64;
    float* rhb    = ws;            ws += N64;
    float* htb    = ws;            ws += N64;

    // Output slices: [prior_mean, prior_std, enc_mean, enc_std, z, h0, h1]
    float* outp = (float*)d_out;
    float* pm  = outp + 0 * N64;
    float* ps  = outp + 1 * N64;
    float* em  = outp + 2 * N64;
    float* es  = outp + 3 * N64;
    float* zz  = outp + 4 * N64;
    float* h0o = outp + 5 * N64;
    float* h1o = outp + 6 * N64;

    const int TB = 256;
    const int nbN   = (N + TB - 1) / TB;
    const int nbE   = (E + TB - 1) / TB;
    const int nbN16 = (int)((N16 + TB - 1) / TB);
    const int nbN24 = (int)(((long)N * 24 + TB - 1) / TB);
    const int nbAgg = (int)(((long)E * 16 + TB - 1) / TB);
    const int GB = (N + 127) / 128;

    // GCN-internal GEMM (no activation, zero bias, single output, stride 64)
    auto GCNGEMM = [&](int KT, const float* A, int Kw, const float* W) {
        if (KT == 128)
            k_gemmT<128, ACT_NONE, false><<<GB, TB, 0, stream>>>(
                A, Kw, W, zero64, tmpHW, 64, 0, nullptr, 0, 0, N);
        else
            k_gemmT<64, ACT_NONE, false><<<GB, TB, 0, stream>>>(
                A, Kw, W, zero64, tmpHW, 64, 0, nullptr, 0, 0, N);
    };
    auto FIN = [&](int act, float* o, const float* b, const float* accum) {
        switch (act) {
        case ACT_RELU:     k_finalize<ACT_RELU>    <<<nbN16, TB, 0, stream>>>(o, agg, tmpHW, norms, b, accum, N); break;
        case ACT_SIGMOID:  k_finalize<ACT_SIGMOID> <<<nbN16, TB, 0, stream>>>(o, agg, tmpHW, norms, b, accum, N); break;
        case ACT_TANH:     k_finalize<ACT_TANH>    <<<nbN16, TB, 0, stream>>>(o, agg, tmpHW, norms, b, accum, N); break;
        case ACT_SOFTPLUS: k_finalize<ACT_SOFTPLUS><<<nbN16, TB, 0, stream>>>(o, agg, tmpHW, norms, b, accum, N); break;
        default:           k_finalize<ACT_NONE>    <<<nbN16, TB, 0, stream>>>(o, agg, tmpHW, norms, b, accum, N); break;
        }
    };
    auto GCN = [&](int KT, const float* A, int Kw, const float* W, const float* b,
                   float* o, const float* accum, int act) {
        GCNGEMM(KT, A, Kw, W);
        hipMemsetAsync(agg, 0, (size_t)N64 * sizeof(float), stream);
        k_agg<<<nbAgg, TB, 0, stream>>>(tmpHW, src, dst, norme, agg, E);
        FIN(act, o, b, accum);
    };

    // Degrees + GCN normalization (once, reused by all 15 GCNs)
    hipMemsetAsync(deg, 0, (size_t)N * sizeof(float), stream);
    hipMemsetAsync(zero64, 0, 64 * sizeof(float), stream);
    k_deg<<<nbE, TB, 0, stream>>>(dst, deg, E);
    k_node_norm<<<nbN, TB, 0, stream>>>(deg, dinv, norms, N);
    k_edge_norm<<<nbE, TB, 0, stream>>>(src, dst, dinv, norme, E);

    // Concat buffers not produced by GEMMs
    k_pack64<<<nbN16, TB, 0, stream>>>(h1, 64, 0, catXH, 128, 64, N);   // catXH[:,64:] = h1
    k_packP<<<nbN24, TB, 0, stream>>>(h1, diff, catP, N);               // [h1|diff|0]

    // phiX -> both concat buffers (cols 0..63)
    k_gemmT<128, ACT_RELU, true><<<GB, TB, 0, stream>>>(
        x, 128, Wpx, bpx, catXH, 128, 0, catXZ, 128, 0, N);

    // Encoder
    GCN(128, catXH, 128, Wen, ben, encx, nullptr, ACT_RELU);            // enc
    GCN(64,  encx,  64,  Wem, bem, em,  nullptr, ACT_NONE);             // enc_mean
    GCN(64,  encx,  64,  Wes, bes, es,  nullptr, ACT_SOFTPLUS);         // enc_std

    // Prior
    k_gemmT<96, ACT_RELU, false><<<GB, TB, 0, stream>>>(
        catP, 65, Wpr, bpr, priorx, 64, 0, nullptr, 0, 0, N);
    k_gemmT<64, ACT_NONE, false><<<GB, TB, 0, stream>>>(
        priorx, 64, Wpm, bpm, pm, 64, 0, nullptr, 0, 0, N);
    k_gemmT<64, ACT_SOFTPLUS, false><<<GB, TB, 0, stream>>>(
        priorx, 64, Wps, bps, ps, 64, 0, nullptr, 0, 0, N);

    // z = eps*enc_std + enc_mean ; phiZ = relu(z @ Wpz + bpz) -> catXZ[:,64:]
    k_zcomb<<<nbN16, TB, 0, stream>>>(eps, es, em, zz, N16);
    k_gemmT<64, ACT_RELU, false><<<GB, TB, 0, stream>>>(
        zz, 64, Wpz, bpz, catXZ, 128, 64, nullptr, 0, 0, N);

    // GRU layers
    const float* hcur[2] = { h0, h1 };
    float*       hout[2] = { h0o, h1o };
    const float* inp = catXZ; int KI = 128;
    for (int l = 0; l < 2; ++l) {
        const float* hi = hcur[l];
        GCN(KI, inp, KI, Wg[l][0], bg[l][0], tmp1, nullptr, ACT_NONE);   // xz
        GCN(64, hi, 64,  Wg[l][1], bg[l][1], zgb, tmp1, ACT_SIGMOID);    // hz -> z gate
        GCN(KI, inp, KI, Wg[l][2], bg[l][2], tmp1, nullptr, ACT_NONE);   // xr
        GCN(64, hi, 64,  Wg[l][3], bg[l][3], rgb, tmp1, ACT_SIGMOID);    // hr -> r gate
        k_mul<<<nbN16, TB, 0, stream>>>(rgb, hi, rhb, N16);              // r*h
        GCN(KI, inp, KI, Wg[l][4], bg[l][4], tmp1, nullptr, ACT_NONE);   // xh
        GCN(64, rhb, 64, Wg[l][5], bg[l][5], htb, tmp1, ACT_TANH);       // hh -> h~
        k_gru<<<nbN16, TB, 0, stream>>>(zgb, hi, htb, hout[l], N16);     // h_out
        inp = hout[l]; KI = 64;
    }
}
